// PytorchFalconAttentionModel_21766894256316
// MI455X (gfx1250) — matrix operand order repaired and checked by reading, not compiled, not run on hardware
//
#include <hip/hip_runtime.h>
#include <cstdint>
#include <cstddef>

// Problem constants (Falcon MQA): B=1, S=1024, HIDDEN=4544, HEADS=71, HEAD_DIM=64
#define S_LEN   1024
#define HID     4544
#define FW      4672      // HIDDEN + 2*HEAD_DIM (fused qkv width)
#define HEADS_N 71
#define HDIM    64
#define KOFF    4544
#define VOFF    4608
#define INV_NORM 0.125f   // 1/sqrt(64)
#define NEG_BIG (-1.0e9f)

#if defined(__AMDGCN__) && __has_builtin(__builtin_amdgcn_global_load_async_to_lds_b128)
#define ASYNC_LDS 1
#else
#define ASYNC_LDS 0
#endif

typedef __attribute__((ext_vector_type(16))) __bf16 v16bf;
typedef __attribute__((ext_vector_type(8)))  float  v8f;
typedef int v4i_vs __attribute__((vector_size(16)));   // matches builtin param type
typedef __attribute__((address_space(1))) v4i_vs* gptr128;
typedef __attribute__((address_space(3))) v4i_vs* lptr128;

union FragBF { uint32_t u[8]; v16bf v; };

__device__ __forceinline__ uint16_t f2bf(float f) {
  uint32_t u = __float_as_uint(f);
  u += 0x7FFFu + ((u >> 16) & 1u);   // round-to-nearest-even
  return (uint16_t)(u >> 16);
}
__device__ __forceinline__ uint32_t pack2bf(float lo, float hi) {
  return (uint32_t)f2bf(lo) | ((uint32_t)f2bf(hi) << 16);
}
// 16-bit A-matrix 16x32 VGPR->K mapping (contiguous, matches B fills):
// lanes 0-15: V0..7 = K{0..15}; lanes 16-31: K{16..31}
__device__ __forceinline__ int a_kidx(int j, int hf) {
  return 16 * hf + 2 * j;
}
__device__ __forceinline__ v8f wmma_bf16(const v16bf& a, const v16bf& b, const v8f& c) {
  return __builtin_amdgcn_wmma_f32_16x16x32_bf16(false, a, false, b, (short)0, c,
                                                 false, false);
}

// ---------------------------------------------------------------------------
// GEMM: C[M,N] = A[M,K] * B[K,N]; A f32 or bf16, B f32 (converted to bf16),
// C f32 or bf16. Block tile 256x64, 8 waves as 4(M)x2(N) -> 64x32 per wave
// (4x2 WMMA tiles, 8 WMMA per 32-deep k-step). Double-buffered LDS.
// ---------------------------------------------------------------------------
template <bool A_BF16, bool OUT_BF16>
__global__ __launch_bounds__(256) void gemm_wmma(const void* __restrict__ A_,
                                                 const float* __restrict__ B_,
                                                 void* __restrict__ C_,
                                                 int M, int N, int K) {
  __shared__ uint16_t sA[2][256][32];   // bf16 [m][k], double buffered (32KB)
  __shared__ uint16_t sBt[2][64][34];   // bf16 [n][k] transposed, padded (8.5KB)

  const int tid  = threadIdx.x;
  const int wave = tid >> 5;
  const int lane = tid & 31;
  const int l16  = lane & 15;
  const int hf   = lane >> 4;
  const int wm   = wave & 3;   // 4 m-waves * 64
  const int wn   = wave >> 2;  // 2 n-waves * 32
  const int m0   = blockIdx.y * 256;
  const int n0   = blockIdx.x * 64;

  const v8f zacc = {0.f, 0.f, 0.f, 0.f, 0.f, 0.f, 0.f, 0.f};
  v8f acc[4][2];
#pragma unroll
  for (int i = 0; i < 4; ++i)
#pragma unroll
    for (int j = 0; j < 2; ++j) acc[i][j] = zacc;

  auto stage = [&](int p, int kk) {
    // A tile 256x32 -> bf16 LDS
    if constexpr (A_BF16) {
      const uint16_t* A = (const uint16_t*)A_;
#if ASYNC_LDS
      for (int i = tid; i < 256 * 4; i += 256) {   // 1024 async B128 copies
        const int m = i >> 2, c8 = (i & 3) << 3;
        __builtin_amdgcn_global_load_async_to_lds_b128(
            (gptr128)&A[(size_t)(m0 + m) * K + kk + c8],
            (lptr128)&sA[p][m][c8], 0, 0);
      }
#else
      for (int i = tid; i < 256 * 8; i += 256) {
        const int m = i >> 3, kc = (i & 7) << 2;
        const uint2 u = *(const uint2*)&A[(size_t)(m0 + m) * K + kk + kc];
        *(uint32_t*)&sA[p][m][kc]     = u.x;
        *(uint32_t*)&sA[p][m][kc + 2] = u.y;
      }
#endif
    } else {
      const float* A = (const float*)A_;
      for (int i = tid; i < 256 * 8; i += 256) {
        const int m = i >> 3, kc = (i & 7) << 2;
        const float4 f = *(const float4*)&A[(size_t)(m0 + m) * K + kk + kc];
        *(uint32_t*)&sA[p][m][kc]     = pack2bf(f.x, f.y);
        *(uint32_t*)&sA[p][m][kc + 2] = pack2bf(f.z, f.w);
      }
    }
    // B tile 32x64 -> bf16 LDS, transposed to [n][k]
    for (int i = tid; i < 32 * 16; i += 256) {
      const int k = i >> 4, nc = (i & 15) << 2;
      const float4 f = *(const float4*)&B_[(size_t)(kk + k) * N + n0 + nc];
      sBt[p][nc + 0][k] = f2bf(f.x);
      sBt[p][nc + 1][k] = f2bf(f.y);
      sBt[p][nc + 2][k] = f2bf(f.z);
      sBt[p][nc + 3][k] = f2bf(f.w);
    }
    // hint the k+2 B tile toward GL2 (weights are the critical stream)
    if (kk + 32 < K && tid < 32)
      __builtin_prefetch((const void*)&B_[(size_t)(kk + 32 + tid) * N + n0], 0, 0);
  };
  auto stage_fence = [&]() {
#if ASYNC_LDS
    if constexpr (A_BF16) asm volatile("s_wait_asynccnt 0" ::: "memory");
#endif
  };

  stage(0, 0);
  stage_fence();
  __syncthreads();

  for (int k0 = 0; k0 < K; k0 += 32) {
    const int p = (k0 >> 5) & 1;
    if (k0 + 32 < K) stage(p ^ 1, k0 + 32);  // overlap next tile with WMMA

    FragBF a[4], b[2];
#pragma unroll
    for (int i = 0; i < 4; ++i) {
      const int m = wm * 64 + i * 16 + l16;
#pragma unroll
      for (int j = 0; j < 8; ++j)
        a[i].u[j] = *(const uint32_t*)&sA[p][m][a_kidx(j, hf)];
    }
#pragma unroll
    for (int j = 0; j < 2; ++j) {
      const int n = wn * 32 + j * 16 + l16;
#pragma unroll
      for (int q = 0; q < 8; ++q)
        b[j].u[q] = *(const uint32_t*)&sBt[p][n][16 * hf + 2 * q];
    }
#pragma unroll
    for (int i = 0; i < 4; ++i)
#pragma unroll
      for (int j = 0; j < 2; ++j)
        acc[i][j] = wmma_bf16(a[i].v, b[j].v, acc[i][j]);

    stage_fence();
    __syncthreads();
  }

  // epilogue: C layout rows r+8*hf, cols l16 (ISA 7.12.2)
#pragma unroll
  for (int i = 0; i < 4; ++i)
#pragma unroll
    for (int j = 0; j < 2; ++j)
#pragma unroll
      for (int r = 0; r < 8; ++r) {
        const int row = m0 + wm * 64 + i * 16 + r + 8 * hf;
        const int col = n0 + wn * 32 + j * 16 + l16;
        if constexpr (OUT_BF16)
          ((uint16_t*)C_)[(size_t)row * N + col] = f2bf(acc[i][j][r]);
        else
          ((float*)C_)[(size_t)row * N + col] = acc[i][j][r];
      }
}

// ---------------------------------------------------------------------------
// Flash attention (multi-query: 1 KV head). One workgroup = (head, 64-query
// block); 4 waves, each owning 16 query rows. Streams 64-key K/V tiles
// through LDS (K via async-to-LDS when available); online softmax; WMMA for
// Q*K^T and P*V.
// ---------------------------------------------------------------------------
__global__ __launch_bounds__(128) void attn_wmma(const uint16_t* __restrict__ fused,
                                                 const float* __restrict__ alibi,
                                                 uint16_t* __restrict__ ctx) {
  __shared__ uint16_t kbuf[64][64];      // [key][d]  (B-frag for Q*K^T)
  __shared__ uint16_t vbufT[64][66];     // [d][key]  (B-frag for P*V)
  __shared__ uint16_t pbuf[4][16][64];   // per-wave P strip [q][key]

  const int head = blockIdx.x;
  const int qb   = blockIdx.y;
  const int tid  = threadIdx.x;
  const int wave = tid >> 5;
  const int lane = tid & 31;
  const int l16  = lane & 15;
  const int hf   = lane >> 4;
  const int qrow = qb * 64 + wave * 16;

  // persistent Q fragments: 2 k-steps over head dim (bf16, from fused buffer)
  FragBF aq[2];
  {
    const uint16_t* qp = fused + (size_t)(qrow + l16) * FW + head * HDIM;
#pragma unroll
    for (int s = 0; s < 2; ++s)
#pragma unroll
      for (int j = 0; j < 8; ++j)
        aq[s].u[j] = *(const uint32_t*)&qp[32 * s + a_kidx(j, hf)];
  }

  const v8f zacc = {0.f, 0.f, 0.f, 0.f, 0.f, 0.f, 0.f, 0.f};
  float mrow[8], lrow[8];
  v8f acc[4];
#pragma unroll
  for (int r = 0; r < 8; ++r) { mrow[r] = -3.0e38f; lrow[r] = 0.0f; }
#pragma unroll
  for (int t = 0; t < 4; ++t) acc[t] = zacc;

  const int kend = qb * 64 + 64;  // causal: no key blocks past the query block
  for (int k0 = 0; k0 < kend; k0 += 64) {
    // K tile: straight copy -> async-to-LDS (ASYNCcnt) when available
#if ASYNC_LDS
    for (int i = tid; i < 64 * 8; i += 128) {     // 512 async B128 copies
      const int key = i >> 3, c8 = (i & 7) << 3;
      __builtin_amdgcn_global_load_async_to_lds_b128(
          (gptr128)&fused[(size_t)(k0 + key) * FW + KOFF + c8],
          (lptr128)&kbuf[key][c8], 0, 0);
    }
#else
    for (int i = tid; i < 64 * 32; i += 128) {
      const int key = i >> 5, c2 = (i & 31) << 1;
      const uint32_t kv = *(const uint32_t*)&fused[(size_t)(k0 + key) * FW + KOFF + c2];
      *(uint32_t*)&kbuf[key][c2] = kv;
    }
#endif
    // V tile: transpose during staging (must go through VGPRs)
    for (int i = tid; i < 64 * 32; i += 128) {
      const int key = i >> 5, c2 = (i & 31) << 1;
      const uint32_t vv = *(const uint32_t*)&fused[(size_t)(k0 + key) * FW + VOFF + c2];
      vbufT[c2 + 0][key] = (uint16_t)(vv & 0xFFFFu);
      vbufT[c2 + 1][key] = (uint16_t)(vv >> 16);
    }
#if ASYNC_LDS
    asm volatile("s_wait_asynccnt 0" ::: "memory");
#endif
    __syncthreads();

    // S = Q * K^T  (4 n-tiles of 16 keys; 2 WMMA k-steps over head dim)
    v8f st[4];
#pragma unroll
    for (int t = 0; t < 4; ++t) {
      FragBF bk;
      const int n = t * 16 + l16;  // key column
      v8f c = zacc;
#pragma unroll
      for (int j = 0; j < 8; ++j)
        bk.u[j] = *(const uint32_t*)&kbuf[n][16 * hf + 2 * j];
      c = wmma_bf16(aq[0].v, bk.v, c);
#pragma unroll
      for (int j = 0; j < 8; ++j)
        bk.u[j] = *(const uint32_t*)&kbuf[n][32 + 16 * hf + 2 * j];
      st[t] = wmma_bf16(aq[1].v, bk.v, c);
    }

    // logits = (S + alibi) * inv_norm + causal mask; online softmax update
    float al[4];
#pragma unroll
    for (int t = 0; t < 4; ++t)
      al[t] = alibi[head * S_LEN + k0 + t * 16 + l16];

#pragma unroll
    for (int r = 0; r < 8; ++r) {
      const int q = qrow + r + 8 * hf;
#pragma unroll
      for (int t = 0; t < 4; ++t) {
        const int kk = k0 + t * 16 + l16;
        const float lg = (st[t][r] + al[t]) * INV_NORM;
        st[t][r] = (kk <= q) ? lg : NEG_BIG;
      }
      float mb = fmaxf(fmaxf(st[0][r], st[1][r]), fmaxf(st[2][r], st[3][r]));
#pragma unroll
      for (int off = 1; off < 16; off <<= 1)
        mb = fmaxf(mb, __shfl_xor(mb, off, 32));   // stays within half-wave
      const float mn    = fmaxf(mrow[r], mb);
      const float scale = __expf(mrow[r] - mn);
      mrow[r] = mn;
      float rs = 0.0f;
#pragma unroll
      for (int t = 0; t < 4; ++t) {
        const float p = __expf(st[t][r] - mn);
        st[t][r] = p;
        rs += p;
      }
#pragma unroll
      for (int off = 1; off < 16; off <<= 1) rs += __shfl_xor(rs, off, 32);
      lrow[r] = lrow[r] * scale + rs;
#pragma unroll
      for (int t = 0; t < 4; ++t) acc[t][r] *= scale;
    }

    // P: C-layout -> A-layout via this wave's private LDS strip
#pragma unroll
    for (int r = 0; r < 8; ++r)
#pragma unroll
      for (int t = 0; t < 4; ++t)
        pbuf[wave][r + 8 * hf][t * 16 + l16] = f2bf(st[t][r]);
    asm volatile("s_wait_dscnt 0" ::: "memory");  // same-wave LDS RAW only

    // ctx += P * V  (2 WMMA k-steps over 64 keys, 4 d-tiles)
#pragma unroll
    for (int s = 0; s < 2; ++s) {
      FragBF pa;
#pragma unroll
      for (int j = 0; j < 8; ++j)
        pa.u[j] = *(const uint32_t*)&pbuf[wave][l16][32 * s + a_kidx(j, hf)];
#pragma unroll
      for (int t = 0; t < 4; ++t) {
        FragBF bv;
        const int n = t * 16 + l16;  // d column
#pragma unroll
        for (int j = 0; j < 8; ++j)
          bv.u[j] = *(const uint32_t*)&vbufT[n][32 * s + 16 * hf + 2 * j];
        acc[t] = wmma_bf16(pa.v, bv.v, acc[t]);
      }
    }
    __syncthreads();  // protect K/V tiles before next staging round
  }

  // normalize by row sums, write ctx as bf16 [1024][4544]
#pragma unroll
  for (int t = 0; t < 4; ++t)
#pragma unroll
    for (int r = 0; r < 8; ++r) {
      const int row = qrow + r + 8 * hf;
      const int col = head * HDIM + t * 16 + l16;
      ctx[(size_t)row * HID + col] = f2bf(acc[t][r] / lrow[r]);
    }
}

// ---------------------------------------------------------------------------
extern "C" void kernel_launch(void* const* d_in, const int* in_sizes, int n_in,
                              void* d_out, int out_size, void* d_ws, size_t ws_size,
                              hipStream_t stream) {
  (void)in_sizes; (void)n_in; (void)out_size; (void)ws_size;
  const float* x      = (const float*)d_in[0];  // [1,1024,4544]
  const float* alibi  = (const float*)d_in[1];  // [1,71,1,1024]
  // d_in[2] attention_mask: causal, computed analytically in-kernel
  const float* Wqkv   = (const float*)d_in[3];  // [4544,4672]
  const float* Wdense = (const float*)d_in[4];  // [4544,4544]
  float* out = (float*)d_out;                   // [1,1024,4544] f32

  uint16_t* fusedbf = (uint16_t*)d_ws;                                   // 1024x4672 bf16
  uint16_t* ctxbf   = (uint16_t*)((char*)d_ws + (size_t)1024 * FW * 2);  // 1024x4544 bf16

  // 1) fused = x @ Wqkv  (f32 in, bf16 out)
  gemm_wmma<false, true><<<dim3(FW / 64, S_LEN / 256), 256, 0, stream>>>(
      (const void*)x, Wqkv, (void*)fusedbf, S_LEN, FW, HID);

  // 2) flash attention with alibi + causal mask -> ctx (bf16)
  attn_wmma<<<dim3(HEADS_N, S_LEN / 64), 128, 0, stream>>>(fusedbf, alibi, ctxbf);

  // 3) out = ctx @ Wdense  (bf16 in, f32 out)
  gemm_wmma<true, false><<<dim3(HID / 64, S_LEN / 256), 256, 0, stream>>>(
      (const void*)ctxbf, Wdense, (void*)out, S_LEN, HID, HID);
}
